// ProbableCollapseLayer_69741678952845
// MI455X (gfx1250) — matrix operand order repaired and checked by reading, not compiled, not run on hardware
//
#include <hip/hip_runtime.h>
#include <hip/hip_bf16.h>

typedef __attribute__((ext_vector_type(16))) __bf16 bf16x16;
typedef __attribute__((ext_vector_type(8)))  __bf16 bf16x8;
typedef __attribute__((ext_vector_type(8)))  float  f32x8;

#define S_    2048
#define D_    1024
#define P_    9
#define DM_   256
#define NTOK  8192
#define EPSF  1e-5f
#define XLDA  1032   // 1024 + 8 bf16 pad (row stride 516 dw -> +4 banks/row)
#define HLDA_BF 264  // 256 + 8 bf16 pad
#define HLDA_F  260  // 256 + 4 f32 pad
#define YLDA  1032   // 1024 + 8 f32 pad (+8 banks/row -> LN groups conflict-free)

// ---- ws byte offsets ----
#define W1SWZ_OFF 0u            // 9*512 tiles *32 lanes *16 bf16 = 4,718,592 B
#define W2SWZ_OFF 4718592u
#define WO1SWZ_OFF 9437184u     // 512*32*16*2 = 524,288 B
#define XM_OFF    9961472u      // 8192*1024*2 = 16,777,216 B
#define CNT_OFF   26738688u     // 9*4 -> pad 256
#define LIST_OFF  26738944u     // 9*8192*4 = 294,912 B

#define WMMA_BF16(a, b, c) \
  __builtin_amdgcn_wmma_f32_16x16x32_bf16(false, (a), false, (b), (short)0, (c), false, false)

__device__ __forceinline__ bf16x16 make_a(const __bf16* rowp) {
  bf16x8 a0 = *(const bf16x8*)rowp;
  bf16x8 a1 = *(const bf16x8*)(rowp + 16);
  return __builtin_shufflevector(a0, a1, 0,1,2,3,4,5,6,7,8,9,10,11,12,13,14,15);
}

// ------------------------------------------------------------------
// Kernel 1: convert + swizzle weights to WMMA B-operand layout, zero counters.
// B fragment: lane l (l<16 -> K=kt*32+{0..7,16..23}, l>=16 -> K=kt*32+{8..15,24..31}),
//             col = nt*16 + (l&15); 16 bf16 stored contiguously per lane.
// ------------------------------------------------------------------
__global__ void prep_kernel(const float* __restrict__ W1,
                            const float* __restrict__ W2,
                            const float* __restrict__ Wo1,
                            __bf16* __restrict__ w1s, __bf16* __restrict__ w2s,
                            __bf16* __restrict__ wo1s, unsigned* __restrict__ cnt) {
  int tid = blockIdx.x * blockDim.x + threadIdx.x;
  if (blockIdx.x == 0 && threadIdx.x < P_) cnt[threadIdx.x] = 0u;
  const int T1 = 9 * 512 * 32;        // W1: 9 paths * (32 kt * 16 nt) tiles * 32 lanes
  const int T2 = T1 + 9 * 512 * 32;   // W2: 9 paths * (8 kt * 64 nt)
  const int T3 = T2 + 512 * 32;       // Wo1: 32 kt * 16 nt
  if (tid >= T3) return;

  const float* src; __bf16* dst; int N, kt, nt, l;
  if (tid < T1) {
    l = tid & 31; int tile = tid >> 5;
    int rem = tile & 511; kt = rem >> 4; nt = rem & 15;
    src = W1 + (size_t)(tile >> 9) * D_ * DM_;
    dst = w1s + ((size_t)tile * 32 + l) * 16;  N = DM_;
  } else if (tid < T2) {
    int local = tid - T1; l = local & 31; int tile = local >> 5;
    int rem = tile & 511; kt = rem >> 6; nt = rem & 63;
    src = W2 + (size_t)(tile >> 9) * DM_ * D_;
    dst = w2s + ((size_t)tile * 32 + l) * 16;  N = D_;
  } else {
    int local = tid - T2; l = local & 31; int tile = local >> 5;
    kt = tile >> 4; nt = tile & 15;
    src = Wo1; dst = wo1s + ((size_t)tile * 32 + l) * 16;  N = DM_;
  }
  int col   = nt * 16 + (l & 15);
  int kbase = kt * 32 + (l >> 4) * 8;
  bf16x16 v;
#pragma unroll
  for (int i = 0; i < 16; ++i) v[i] = (__bf16)src[(size_t)(kbase + i + (i >= 8 ? 8 : 0)) * N + col];
  *(bf16x16*)dst = v;
}

// ------------------------------------------------------------------
// Kernel 2: 3-tap mix -> bf16 x_mixed (LDS + global), observer WMMA GEMM
//           (4 accumulators per wave), tanh, tiny GEMM2 to 9 scores,
//           argmax, append token to per-path list.
// ------------------------------------------------------------------
__global__ void __launch_bounds__(128)
mix_observe_kernel(const float* __restrict__ x, const float* __restrict__ mixk,
                   const float* __restrict__ bo1, const float* __restrict__ Wo2,
                   const float* __restrict__ bo2, const __bf16* __restrict__ wo1s,
                   __bf16* __restrict__ xm, unsigned* __restrict__ cnt,
                   int* __restrict__ lists) {
  __shared__ __bf16 Xs[16][XLDA];
  __shared__ float  Hs[16][HLDA_F];
  __shared__ float  Sc[16][P_];
  const int tid = threadIdx.x;
  const int tok0 = blockIdx.x * 16;

  // depthwise 3-tap mix along sequence (zero pad at batch edges)
  for (int r = 0; r < 16; ++r) {
    int tok = tok0 + r;
    int s = tok & (S_ - 1);
    const float* xr  = x + (size_t)tok * D_;
    const float* xm1 = (s > 0)      ? xr - D_ : nullptr;
    const float* xp1 = (s < S_ - 1) ? xr + D_ : nullptr;
    __bf16* dstg = xm + (size_t)tok * D_;
#pragma unroll
    for (int j = 0; j < 8; ++j) {
      int d = tid + j * 128;
      float a = xm1 ? xm1[d] : 0.f;
      float b = xr[d];
      float c = xp1 ? xp1[d] : 0.f;
      float v = mixk[d] * a + mixk[D_ + d] * b + mixk[2 * D_ + d] * c;
      __bf16 bv = (__bf16)v;
      Xs[r][d] = bv;
      dstg[d]  = bv;
    }
  }
  __syncthreads();

  // observer GEMM1: (16x1024) @ (1024x256) -> tanh -> Hs
  // wave handles 4 consecutive N-tiles: A fragment reused by 4 WMMAs.
  const int wave = tid >> 5, l = tid & 31;
  const int rowA = l & 15, hi = l >> 4;
  {
    const int nt0 = wave * 4;
    f32x8 c0 = {}, c1 = {}, c2 = {}, c3 = {};
    for (int kt = 0; kt < 32; ++kt) {
      bf16x16 a = make_a(&Xs[rowA][kt * 32 + hi * 8]);
      const bf16x16* bp = (const bf16x16*)wo1s + ((size_t)(kt * 16 + nt0) * 32 + l);
      if (kt < 31) __builtin_prefetch((const void*)(bp + 16 * 32), 0, 0);
      bf16x16 b0 = bp[0 * 32], b1v = bp[1 * 32], b2v = bp[2 * 32], b3v = bp[3 * 32];
      c0 = WMMA_BF16(a, b0,  c0);
      c1 = WMMA_BF16(a, b1v, c1);
      c2 = WMMA_BF16(a, b2v, c2);
      c3 = WMMA_BF16(a, b3v, c3);
    }
    f32x8 cs[4] = {c0, c1, c2, c3};
#pragma unroll
    for (int j = 0; j < 4; ++j) {
      int colC = (nt0 + j) * 16 + (l & 15);
      float bo = bo1[colC];
#pragma unroll
      for (int r = 0; r < 8; ++r) {
        int rowC = (l < 16) ? r : r + 8;
        Hs[rowC][colC] = tanhf(cs[j][r] + bo);
      }
    }
  }
  __syncthreads();

  // GEMM2: (16x256) @ (256x9), scalar (tiny)
  if (tid < 16 * P_) {
    int r = tid / P_, p = tid % P_;
    float s = bo2[p];
    for (int m = 0; m < DM_; ++m) s += Hs[r][m] * Wo2[m * P_ + p];
    Sc[r][p] = s;
  }
  __syncthreads();

  if (tid < 16) {
    int best = 0; float bv = Sc[tid][0];
#pragma unroll
    for (int p = 1; p < P_; ++p) { float v = Sc[tid][p]; if (v > bv) { bv = v; best = p; } }
    unsigned pos = atomicAdd(&cnt[best], 1u);
    lists[best * NTOK + pos] = tok0 + tid;
  }
}

// ------------------------------------------------------------------
// Kernel 3: per-path gathered bottleneck (two WMMA GEMMs, 4 accumulators
//           per wave) + fused residual LayerNorm writing final output.
// ------------------------------------------------------------------
__global__ void __launch_bounds__(128)
path_gemm_ln_kernel(const float* __restrict__ x, const float* __restrict__ b1,
                    const float* __restrict__ b2, const float* __restrict__ gamma,
                    const float* __restrict__ beta, const __bf16* __restrict__ w1s,
                    const __bf16* __restrict__ w2s, const __bf16* __restrict__ xm,
                    const unsigned* __restrict__ cnt, const int* __restrict__ lists,
                    float* __restrict__ out) {
  __shared__ __bf16 Xs[16][XLDA];
  __shared__ __bf16 Hb[16][HLDA_BF];
  __shared__ float  Ys[16][YLDA];
  __shared__ int    toks[16];

  const int p    = blockIdx.x >> 9;
  const int tile = blockIdx.x & 511;
  const int n    = (int)cnt[p];
  const int base = tile * 16;
  if (base >= n) return;
  const int nvalid = min(16, n - base);
  const int tid = threadIdx.x;

  if (tid < 16) {
    int idx = base + tid;
    toks[tid] = lists[p * NTOK + min(idx, n - 1)];
  }
  __syncthreads();

  // gather 16 x_mixed rows (bf16) into LDS: 128 threads * 16B = one row
  for (int r = 0; r < 16; ++r) {
    const uint4* src = (const uint4*)(xm + (size_t)toks[r] * D_);
    ((uint4*)&Xs[r][0])[tid] = src[tid];
  }
  __syncthreads();

  const int wave = tid >> 5, l = tid & 31;
  const int rowA = l & 15, hi = l >> 4;

  // GEMM1: (16x1024) @ W1[p](1024x256) + b1 -> bf16 Hb
  const __bf16* w1p = w1s + (size_t)p * 512 * 32 * 16;
  {
    const int nt0 = wave * 4;
    f32x8 c0 = {}, c1 = {}, c2 = {}, c3 = {};
    for (int kt = 0; kt < 32; ++kt) {
      bf16x16 a = make_a(&Xs[rowA][kt * 32 + hi * 8]);
      const bf16x16* bp = (const bf16x16*)w1p + ((size_t)(kt * 16 + nt0) * 32 + l);
      if (kt < 31) __builtin_prefetch((const void*)(bp + 16 * 32), 0, 0);
      bf16x16 b0 = bp[0 * 32], b1v = bp[1 * 32], b2v = bp[2 * 32], b3v = bp[3 * 32];
      c0 = WMMA_BF16(a, b0,  c0);
      c1 = WMMA_BF16(a, b1v, c1);
      c2 = WMMA_BF16(a, b2v, c2);
      c3 = WMMA_BF16(a, b3v, c3);
    }
    f32x8 cs[4] = {c0, c1, c2, c3};
#pragma unroll
    for (int j = 0; j < 4; ++j) {
      int colC = (nt0 + j) * 16 + (l & 15);
      float bb = b1[p * DM_ + colC];
#pragma unroll
      for (int r = 0; r < 8; ++r) {
        int rowC = (l < 16) ? r : r + 8;
        Hb[rowC][colC] = (__bf16)(cs[j][r] + bb);
      }
    }
  }
  __syncthreads();

  // GEMM2: (16x256) @ W2[p](256x1024) + b2 -> f32 Ys
  const __bf16* w2p = w2s + (size_t)p * 512 * 32 * 16;
  for (int blk = 0; blk < 4; ++blk) {
    const int nt0 = (blk * 4 + wave) * 4;
    f32x8 c0 = {}, c1 = {}, c2 = {}, c3 = {};
#pragma unroll
    for (int kt = 0; kt < 8; ++kt) {
      bf16x16 a = make_a(&Hb[rowA][kt * 32 + hi * 8]);
      const bf16x16* bp = (const bf16x16*)w2p + ((size_t)(kt * 64 + nt0) * 32 + l);
      if (kt < 7) __builtin_prefetch((const void*)(bp + 64 * 32), 0, 0);
      bf16x16 b0 = bp[0 * 32], b1v = bp[1 * 32], b2v = bp[2 * 32], b3v = bp[3 * 32];
      c0 = WMMA_BF16(a, b0,  c0);
      c1 = WMMA_BF16(a, b1v, c1);
      c2 = WMMA_BF16(a, b2v, c2);
      c3 = WMMA_BF16(a, b3v, c3);
    }
    f32x8 cs[4] = {c0, c1, c2, c3};
#pragma unroll
    for (int j = 0; j < 4; ++j) {
      int colC = (nt0 + j) * 16 + (l & 15);
      float bb = b2[p * D_ + colC];
#pragma unroll
      for (int r = 0; r < 8; ++r) {
        int rowC = (l < 16) ? r : r + 8;
        Ys[rowC][colC] = cs[j][r] + bb;
      }
    }
  }
  __syncthreads();

  // Fused LayerNorm(x + Y): 16 groups of 8 lanes
  const int g = tid >> 3, lane8 = tid & 7;
  if (g < nvalid) {
    int tok = toks[g];
    const float* xr = x + (size_t)tok * D_;
    float s = 0.f, ss = 0.f;
    for (int d = lane8; d < D_; d += 8) {
      float v = xr[d] + Ys[g][d];
      s += v; ss += v * v;
    }
#pragma unroll
    for (int m = 1; m < 8; m <<= 1) {
      s  += __shfl_xor(s, m, 32);
      ss += __shfl_xor(ss, m, 32);
    }
    float mu  = s * (1.f / D_);
    float var = ss * (1.f / D_) - mu * mu;
    float inv = rsqrtf(var + EPSF);
    float* orow = out + (size_t)tok * D_;
    for (int d = lane8; d < D_; d += 8) {
      float v = xr[d] + Ys[g][d];
      orow[d] = (v - mu) * inv * gamma[d] + beta[d];
    }
  }
}

extern "C" void kernel_launch(void* const* d_in, const int* in_sizes, int n_in,
                              void* d_out, int out_size, void* d_ws, size_t ws_size,
                              hipStream_t stream) {
  (void)in_sizes; (void)n_in; (void)out_size; (void)ws_size;
  const float* x    = (const float*)d_in[0];
  const float* mixk = (const float*)d_in[1];
  const float* W1   = (const float*)d_in[2];
  const float* b1   = (const float*)d_in[3];
  const float* W2   = (const float*)d_in[4];
  const float* b2   = (const float*)d_in[5];
  const float* Wo1  = (const float*)d_in[6];
  const float* bo1  = (const float*)d_in[7];
  const float* Wo2  = (const float*)d_in[8];
  const float* bo2  = (const float*)d_in[9];
  const float* gamma= (const float*)d_in[10];
  const float* beta = (const float*)d_in[11];
  float* out = (float*)d_out;

  char* ws = (char*)d_ws;
  __bf16*   w1s   = (__bf16*)(ws + W1SWZ_OFF);
  __bf16*   w2s   = (__bf16*)(ws + W2SWZ_OFF);
  __bf16*   wo1s  = (__bf16*)(ws + WO1SWZ_OFF);
  __bf16*   xmix  = (__bf16*)(ws + XM_OFF);
  unsigned* cnt   = (unsigned*)(ws + CNT_OFF);
  int*      lists = (int*)(ws + LIST_OFF);

  prep_kernel<<<1216, 256, 0, stream>>>(W1, W2, Wo1, w1s, w2s, wo1s, cnt);
  mix_observe_kernel<<<512, 128, 0, stream>>>(x, mixk, bo1, Wo2, bo2, wo1s, xmix,
                                              cnt, lists);
  path_gemm_ln_kernel<<<P_ * 512, 128, 0, stream>>>(x, b1, b2, gamma, beta, w1s, w2s,
                                                    xmix, cnt, lists, out);
}
